// S4R_1803886265844
// MI455X (gfx1250) — matrix operand order repaired and checked by reading, not compiled, not run on hardware
//
#include <hip/hip_runtime.h>

// MI455X / gfx1250: wave32, WMMA f32_16x16x32_f16, LDS-staged implicit-GEMM conv.

typedef __attribute__((ext_vector_type(16))) _Float16 v16h;
typedef __attribute__((ext_vector_type(8)))  float    v8f;

#define Bn 8
#define Hn 512
#define Ln 8192

// ---- conv kernel tiling ----
#define ROWS   1024              // output rows (l) per workgroup
#define NCHUNK (ROWS / 16)       // 64 chunks of 16 rows
#define SCH    1024              // s-values staged per superchunk
#define UROW   (SCH + 64)        // staged row incl. front+back halo
#define KREVLEN (Ln + 224)       // reversed K + zero guards both ends

union AFrag { v16h v; uint32_t u[8]; _Float16 h[16]; };
union BFrag { v16h v; uint4 q[2]; };
union HPack { uint4 q; _Float16 h[8]; };
union H4    { uint2 d; _Float16 h[4]; };

// y[b,h,l] = sum_{t=0..l} K[h,t] u[b,h,l-t] + D[h] u[b,h,l]   -> f16 workspace
// Implicit Toeplitz GEMM: D(16x16) = A(16x32: K window) x B(32x16: u),
// N packs 8 batches x 2 output chunks sharing one Toeplitz block.
__global__ __launch_bounds__(256) void conv_kernel(
    const float* __restrict__ u, const float* __restrict__ Kc,
    const float* __restrict__ D, _Float16* __restrict__ yws)
{
  __shared__ __attribute__((aligned(16))) _Float16 krevA[KREVLEN];
  __shared__ __attribute__((aligned(16))) _Float16 krevB[KREVLEN]; // krevA shifted by one half
  __shared__ __attribute__((aligned(16))) _Float16 uls[Bn][UROW];

  const int h    = blockIdx.y;
  const int blk  = blockIdx.x;
  const int tid  = threadIdx.x;
  // wave id as an SGPR so loop bounds below are provably wave-uniform
  const int wave = __builtin_amdgcn_readfirstlane(tid >> 5);
  const int lane = tid & 31;
  const int mlan = lane & 15;
  const int hi16 = lane >> 4;       // 0 / 1
  const int klo  = hi16 * 8;        // A-fragment k offset (ISA 16-bit A layout)

  // ---- stage reversed K (f16) with zero guards, two byte-shifted copies ----
  const float* Kh = Kc + (size_t)h * Ln;
  for (int i = tid; i < KREVLEN; i += 256) {
    int tA = Ln + 31 - i;   // krevA[i] <-> K[tA]
    int tB = Ln + 30 - i;   // krevB[i] = krevA[i+1]
    krevA[i] = (tA >= 0 && tA < Ln) ? (_Float16)Kh[tA] : (_Float16)0.0f;
    krevB[i] = (tB >= 0 && tB < Ln) ? (_Float16)Kh[tB] : (_Float16)0.0f;
  }

  // wave owns local chunks {8w..8w+7}; pairs (i1, i1+2)  (all SGPR values)
  const int c0 = blk * NCHUNK;
  const int i1a0 = c0 + 8 * wave + 0;
  const int i1a1 = c0 + 8 * wave + 1;
  const int i1a2 = c0 + 8 * wave + 4;
  const int i1a3 = c0 + 8 * wave + 5;
  const int i2max = c0 + 8 * wave + 7;
  const int jend  = (16 * i2max + 15) >> 5;              // wave-uniform (SGPR)
  const int jendB = (16 * (c0 + NCHUNK - 1) + 15) >> 5;  // block-uniform
  const int nsc   = jendB / 32 + 1;

  v8f acc0, acc1, acc2, acc3;
  {
    v8f z = {0.f, 0.f, 0.f, 0.f, 0.f, 0.f, 0.f, 0.f};
    acc0 = z; acc1 = z; acc2 = z; acc3 = z;
  }

  const int nB  = lane & 15;        // WMMA column
  const int bB  = nB & 7;           // batch
  const int jjB = nB >> 3;          // 0 -> U_j, 1 -> U_{j+1}
  const int khB = hi16 * 8;         // B-fragment k offset (ISA 16-bit B layout)

  // j-invariant parity of the A-fragment base -> fixed krevA/krevB select
  const int parc  = ((Ln + 31) - mlan + klo) & 1;
  const uint32_t* kbase =
      (const uint32_t*)(parc ? krevB : krevA);
  // word index at (j=0, i1=0): advances +16 words per j, -8 words per i1
  const int w00 = (((Ln + 31) - mlan + klo) - parc) >> 1;

  for (int sc = 0; sc < nsc; ++sc) {
    __syncthreads();
    const int sbase = sc * SCH;
    // stage u (f16, vectorized): uls[b][q] <-> s = sbase + q - 32 ; zero outside [0,L)
    for (int idx = tid; idx < Bn * (UROW / 4); idx += 256) {
      int b  = idx / (UROW / 4);
      int q  = (idx - b * (UROW / 4)) * 4;
      int s  = sbase + q - 32;                 // multiple of 4
      float4 v = {0.f, 0.f, 0.f, 0.f};
      if (s >= 0 && s < Ln)
        v = *(const float4*)&u[((size_t)b * Hn + h) * Ln + s];
      H4 o;
      o.h[0] = (_Float16)v.x; o.h[1] = (_Float16)v.y;
      o.h[2] = (_Float16)v.z; o.h[3] = (_Float16)v.w;
      *(uint2*)&uls[b][q] = o.d;
    }
    // prefetch next superchunk while this one is consumed (global_prefetch_b8)
    if (sc + 1 < nsc) {
      int b  = tid & 7;
      int s  = sbase + SCH + (tid >> 3) * 32;  // 32 floats = 128B per thread
      if (s < Ln)
        __builtin_prefetch(&u[((size_t)b * Hn + h) * Ln + s], 0, 1);
    }
    __syncthreads();

    const int jlo    = sc * 32;
    const int jstart = (sc == 0) ? -1 : jlo;   // j=-1 feeds s in [0,32) to i+2 columns
    int jhi = jlo + 31; if (jhi > jend) jhi = jend;

    // strength-reduced per-lane addresses (advance +16 words / +32 halves per j)
    int w0 = w00 + 16 * jstart - 8 * i1a0;
    int w1 = w00 + 16 * jstart - 8 * i1a1;
    int w2 = w00 + 16 * jstart - 8 * i1a2;
    int w3 = w00 + 16 * jstart - 8 * i1a3;
    const _Float16* ubp =
        &uls[0][0] + bB * UROW + (32 * (jstart + jjB) + khB - sbase + 32);

#pragma unroll 2
    for (int j = jstart; j <= jhi; ++j) {
      // issue ALL loads first (B + 4 A fragments), then the 4 WMMAs:
      // lets the compiler stagger s_wait_dscnt and overlap across iterations.
      BFrag bf;
      bf.q[0] = *(const uint4*)(ubp);
      bf.q[1] = *(const uint4*)(ubp + 16);
      AFrag a0, a1, a2, a3;
      a0.u[0] = kbase[w0 + 0];  a0.u[1] = kbase[w0 + 1];
      a0.u[2] = kbase[w0 + 2];  a0.u[3] = kbase[w0 + 3];
      a0.u[4] = kbase[w0 + 8];  a0.u[5] = kbase[w0 + 9];
      a0.u[6] = kbase[w0 + 10]; a0.u[7] = kbase[w0 + 11];
      a1.u[0] = kbase[w1 + 0];  a1.u[1] = kbase[w1 + 1];
      a1.u[2] = kbase[w1 + 2];  a1.u[3] = kbase[w1 + 3];
      a1.u[4] = kbase[w1 + 8];  a1.u[5] = kbase[w1 + 9];
      a1.u[6] = kbase[w1 + 10]; a1.u[7] = kbase[w1 + 11];
      a2.u[0] = kbase[w2 + 0];  a2.u[1] = kbase[w2 + 1];
      a2.u[2] = kbase[w2 + 2];  a2.u[3] = kbase[w2 + 3];
      a2.u[4] = kbase[w2 + 8];  a2.u[5] = kbase[w2 + 9];
      a2.u[6] = kbase[w2 + 10]; a2.u[7] = kbase[w2 + 11];
      a3.u[0] = kbase[w3 + 0];  a3.u[1] = kbase[w3 + 1];
      a3.u[2] = kbase[w3 + 2];  a3.u[3] = kbase[w3 + 3];
      a3.u[4] = kbase[w3 + 8];  a3.u[5] = kbase[w3 + 9];
      a3.u[6] = kbase[w3 + 10]; a3.u[7] = kbase[w3 + 11];

      acc0 = __builtin_amdgcn_wmma_f32_16x16x32_f16(
          false, a0.v, false, bf.v, (short)0, acc0, false, false);
      acc1 = __builtin_amdgcn_wmma_f32_16x16x32_f16(
          false, a1.v, false, bf.v, (short)0, acc1, false, false);
      acc2 = __builtin_amdgcn_wmma_f32_16x16x32_f16(
          false, a2.v, false, bf.v, (short)0, acc2, false, false);
      acc3 = __builtin_amdgcn_wmma_f32_16x16x32_f16(
          false, a3.v, false, bf.v, (short)0, acc3, false, false);

      w0 += 16; w1 += 16; w2 += 16; w3 += 16;
      ubp += 32;
    }
  }

  // ---- epilogue: add D*u skip, store f16 y to workspace (16B packed) ----
  const float Dh = D[h];
  const int csel = nB >> 3;          // column 0-7 -> i1, 8-15 -> i2 = i1+2
#pragma unroll
  for (int p = 0; p < 4; ++p) {
    int i1 = (p == 0) ? i1a0 : (p == 1) ? i1a1 : (p == 2) ? i1a2 : i1a3;
    v8f  c = (p == 0) ? acc0 : (p == 1) ? acc1 : (p == 2) ? acc2 : acc3;
    int iabs  = i1 + 2 * csel;
    int lbase = 16 * iabs + hi16 * 8;           // rows m = r + 8*hi16
    const float4* up =
        (const float4*)&u[((size_t)bB * Hn + h) * Ln + lbase];
    float4 u0 = up[0], u1 = up[1];
    HPack o;
    o.h[0] = (_Float16)(c[0] + Dh * u0.x);
    o.h[1] = (_Float16)(c[1] + Dh * u0.y);
    o.h[2] = (_Float16)(c[2] + Dh * u0.z);
    o.h[3] = (_Float16)(c[3] + Dh * u0.w);
    o.h[4] = (_Float16)(c[4] + Dh * u1.x);
    o.h[5] = (_Float16)(c[5] + Dh * u1.y);
    o.h[6] = (_Float16)(c[6] + Dh * u1.z);
    o.h[7] = (_Float16)(c[7] + Dh * u1.w);
    *(uint4*)&yws[((size_t)bB * Hn + h) * Ln + lbase] = o.q;
  }
}

// ---- mix kernel: out[b,p,l] = tanh( sum_h W[p,h] * y[b,h,l] ) ----
#define LT 128     // l tile
#define PT 128     // p tile (8 waves x 16)
#define YP 40      // LDS pitch in halves (16B-aligned rows, conflict-free)

__global__ __launch_bounds__(256) void mix_kernel(
    const _Float16* __restrict__ yws, const float* __restrict__ W,
    float* __restrict__ out)
{
  __shared__ __attribute__((aligned(16))) _Float16 ylds[LT][YP];

  const int b    = blockIdx.z;
  const int pt   = blockIdx.y;
  const int lt   = blockIdx.x;
  const int tid  = threadIdx.x;
  const int wave = __builtin_amdgcn_readfirstlane(tid >> 5);
  const int lane = tid & 31;
  const int mlan = lane & 15;
  const int hi16 = lane >> 4;
  const int p0   = pt * PT + wave * 16;
  const int l0   = lt * LT;

  v8f acc[8];
  {
    v8f z = {0.f, 0.f, 0.f, 0.f, 0.f, 0.f, 0.f, 0.f};
#pragma unroll
    for (int i = 0; i < 8; ++i) acc[i] = z;
  }

  for (int kh = 0; kh < Hn; kh += 32) {
    __syncthreads();
    // stage y transposed: ylds[l][k] = y[b][kh+k][l0+l]
    {
      int kk   = tid >> 3;          // 0..31
      int lseg = (tid & 7) * 16;    // 0..112
      const uint4* src =
          (const uint4*)&yws[((size_t)b * Hn + kh + kk) * Ln + l0 + lseg];
      HPack v0, v1; v0.q = src[0]; v1.q = src[1];
#pragma unroll
      for (int q = 0; q < 8; ++q) ylds[lseg + q][kk] = v0.h[q];
#pragma unroll
      for (int q = 0; q < 8; ++q) ylds[lseg + 8 + q][kk] = v1.h[q];
    }
    __syncthreads();

    // A fragment from W (f32 -> f16 on the fly; rows are h-contiguous)
    AFrag af;
    {
      const float4* wr =
          (const float4*)&W[(size_t)(p0 + mlan) * Hn + kh + hi16 * 8];
      float4 w0 = wr[0], w1 = wr[1], w2 = wr[4], w3 = wr[5];
      af.h[0]  = (_Float16)w0.x; af.h[1]  = (_Float16)w0.y;
      af.h[2]  = (_Float16)w0.z; af.h[3]  = (_Float16)w0.w;
      af.h[4]  = (_Float16)w1.x; af.h[5]  = (_Float16)w1.y;
      af.h[6]  = (_Float16)w1.z; af.h[7]  = (_Float16)w1.w;
      af.h[8]  = (_Float16)w2.x; af.h[9]  = (_Float16)w2.y;
      af.h[10] = (_Float16)w2.z; af.h[11] = (_Float16)w2.w;
      af.h[12] = (_Float16)w3.x; af.h[13] = (_Float16)w3.y;
      af.h[14] = (_Float16)w3.z; af.h[15] = (_Float16)w3.w;
    }
#pragma unroll
    for (int ls = 0; ls < 8; ++ls) {
      BFrag bf;
      const uint4* yp = (const uint4*)&ylds[ls * 16 + mlan][hi16 * 8];
      bf.q[0] = yp[0];
      bf.q[1] = yp[2];
      acc[ls] = __builtin_amdgcn_wmma_f32_16x16x32_f16(
          false, af.v, false, bf.v, (short)0, acc[ls], false, false);
    }
  }

  // epilogue: tanh, coalesced f32 stores (lanes cover 16 consecutive l)
#pragma unroll
  for (int ls = 0; ls < 8; ++ls) {
    int l = l0 + ls * 16 + mlan;
#pragma unroll
    for (int r = 0; r < 8; ++r) {
      int p = p0 + r + hi16 * 8;
      out[((size_t)b * Hn + p) * Ln + l] = tanhf(acc[ls][r]);
    }
  }
}

extern "C" void kernel_launch(void* const* d_in, const int* in_sizes, int n_in,
                              void* d_out, int out_size, void* d_ws, size_t ws_size,
                              hipStream_t stream) {
  const float* u  = (const float*)d_in[0];
  const float* Kc = (const float*)d_in[1];
  const float* D  = (const float*)d_in[2];
  const float* W  = (const float*)d_in[3];
  float*     outp = (float*)d_out;
  _Float16*   yws = (_Float16*)d_ws;   // needs B*H*L*2 = 64 MiB scratch

  conv_kernel<<<dim3(Ln / ROWS, Hn), 256, 0, stream>>>(u, Kc, D, yws);
  mix_kernel<<<dim3(Ln / LT, Hn / PT, Bn), 256, 0, stream>>>(yws, W, outp);
}
